// Attention_32195074851105
// MI455X (gfx1250) — matrix operand order repaired and checked by reading, not compiled, not run on hardware
//
#include <hip/hip_runtime.h>
#include <hip/hip_bf16.h>

// ---------------------------------------------------------------------------
// CDNA5 / gfx1250 implementation (wave32, WMMA, async global->LDS staging).
// Pipeline:
//   0) pad/convert f32 weights + gating inputs to bf16 (zero-padded tiles)
//   1) embed + im2col(LDS) + conv as WMMA GEMM + lrelu -> c (N x 4096) bf16
//      (conv output K padded oc*128+ow; matching W1p columns are zero)
//   2) FC1  : h1  = lrelu(c  @ W1p^T + b1)  WMMA, B async-staged via LDS
//   3) FC2  : hfc = lrelu(h1 @ W2^T + b2)   WMMA
//   4) gate : g = tanh(ld_gcn @ Wg^T + bg), e = tanh(ld_enc @ We^T + be)  WMMA
//   5) rowwise dots -> softmax(2) -> scale ld_gcn / ld_encoder
// ---------------------------------------------------------------------------

#define USE_ASYNC_LDS 1   // gfx1250 GLOBAL_LOAD_ASYNC_TO_LDS_B128 staging

typedef __attribute__((ext_vector_type(16))) __bf16 v16bf;
typedef __attribute__((ext_vector_type(8)))  __bf16 v8bf;
typedef __attribute__((ext_vector_type(8)))  float  v8f;

#define EMB_COLS   1140
#define CONV_OW    124
#define CONV_KP    4096     // 32 oc * 128 (ow padded 124->128)
#define FC1_NP     1024     // 1000 padded to 64*16
#define FC2_NP     112      // 100 padded to 7*16
#define GATE_K     512

__device__ __forceinline__ unsigned short f2bf(float f) {
    union { float f; unsigned int u; } c; c.f = f;
    unsigned int u = c.u;
    u += 0x7FFFu + ((u >> 16) & 1u);       // round-to-nearest-even
    return (unsigned short)(u >> 16);
}

#if USE_ASYNC_LDS
// Async DMA one 16B chunk per lane: global -> LDS. Tracked by ASYNCcnt.
__device__ __forceinline__ void async_copy_b128(unsigned lds_byte_addr,
                                                const void* gsrc) {
    asm volatile("global_load_async_to_lds_b128 %0, %1, off"
                 :: "v"(lds_byte_addr), "v"(gsrc) : "memory");
}
template <int N>
__device__ __forceinline__ void wait_asynccnt() {
    asm volatile("s_wait_asynccnt %0" :: "n"(N) : "memory");
}
// low 32 bits of a generic pointer to LDS == LDS byte offset (aperture rule)
__device__ __forceinline__ unsigned lds_addr_of(const void* p) {
    return (unsigned)(unsigned long long)p;
}
#endif

// ---------------------------------------------------------------------------
// 0a) generic pad + f32->bf16: dst (dR x dC) from src (sR x sC), zero outside.
// ---------------------------------------------------------------------------
__global__ void pad_bf16_kernel(unsigned short* __restrict__ dst,
                                const float* __restrict__ src,
                                int dR, int dC, int sR, int sC) {
    int i = blockIdx.x * blockDim.x + threadIdx.x;
    int total = dR * dC;
    if (i >= total) return;
    int r = i / dC, c = i - r * dC;
    float v = (r < sR && c < sC) ? src[r * sC + c] : 0.0f;
    dst[i] = f2bf(v);
}

// ---------------------------------------------------------------------------
// 0b) W1 remap: dst(1024 x 4096), dst[j][oc*128+ow] = W1[j][oc*124+ow]
//     (zeros for j>=1000, ow>=124) -- matches conv's padded K layout.
// ---------------------------------------------------------------------------
__global__ void pad_w1_kernel(unsigned short* __restrict__ dst,
                              const float* __restrict__ src) {
    int i = blockIdx.x * blockDim.x + threadIdx.x;
    if (i >= FC1_NP * CONV_KP) return;
    int j  = i >> 12;          // / 4096
    int k  = i & 4095;
    int oc = k >> 7, ow = k & 127;
    float v = (j < 1000 && ow < CONV_OW)
                ? src[(size_t)j * 3968 + oc * CONV_OW + ow] : 0.0f;
    dst[i] = f2bf(v);
}

// ---------------------------------------------------------------------------
// 1) embedding gather + im2col in LDS + conv as WMMA + bias + lrelu -> bf16
//    One block per sample. A: 128(ow,pad) x 64(K=2*25 pad) bf16 in LDS.
//    Wcp: 32(oc) x 64 bf16 prepacked. Output c[n][oc*128+ow]; ow>=124 lanes
//    store garbage that is multiplied by zero W1p columns downstream.
// ---------------------------------------------------------------------------
__global__ __launch_bounds__(256)
void conv_embed_wmma(const int* __restrict__ x, const int* __restrict__ y,
                     const float* __restrict__ H,
                     const unsigned short* __restrict__ Wcp,  // 32x64 bf16
                     const float* __restrict__ cb,
                     unsigned short* __restrict__ cbf) {
    __shared__ unsigned short im[128 * 64];   // 16KB im2col tile
    __shared__ unsigned short wsh[32 * 64];   // 4KB weights
    const int n = blockIdx.x;
    const int t = threadIdx.x;
    const float* r0 = H + (size_t)x[n] * EMB_COLS;
    const float* r1 = H + (size_t)(y[n] + 240) * EMB_COLS;

    for (int idx = t; idx < 128 * 64; idx += 256) {
        const int ow = idx >> 6, j = idx & 63;
        float v = 0.0f;
        if (ow < CONV_OW) {
            if (j < 25)      v = r0[ow * 9 + j];
            else if (j < 50) v = r1[ow * 9 + j - 25];
        }
        im[idx] = f2bf(v);
    }
    for (int idx = t; idx < 32 * 64; idx += 256) wsh[idx] = Wcp[idx];
    __syncthreads();

    const int wave = t >> 5, lane = t & 31;
    const int m = lane & 15, half = lane >> 4;
    union AFrag { v16bf v; v8bf h[2]; };

    // 16 output tiles (8 ow-tiles x 2 oc-tiles), 2 per wave.
#pragma unroll
    for (int i = 0; i < 2; ++i) {
        const int ti = wave * 2 + i;
        const int owt = ti >> 1, oct = ti & 1;
        v8f acc = (v8f)(0.0f);
#pragma unroll
        for (int kk = 0; kk < 64; kk += 32) {
            AFrag a;
            const int abase = (owt * 16 + m) * 64 + kk + half * 16;
            a.h[0] = *reinterpret_cast<const v8bf*>(&im[abase]);
            a.h[1] = *reinterpret_cast<const v8bf*>(&im[abase + 8]);
            const v16bf b = *reinterpret_cast<const v16bf*>(
                &wsh[(oct * 16 + m) * 64 + kk + half * 16]);
            acc = __builtin_amdgcn_wmma_f32_16x16x32_bf16(
                false, a.v, false, b, (short)0, acc, false, false);
        }
        // lane's 8 rows are consecutive ow -> one packed 16B store
        const int oc = oct * 16 + m;
        const float bv = cb[oc];
        unsigned int pk[4];
#pragma unroll
        for (int r = 0; r < 8; r += 2) {
            float v0 = acc[r] + bv;     v0 = v0 > 0.0f ? v0 : 0.01f * v0;
            float v1 = acc[r + 1] + bv; v1 = v1 > 0.0f ? v1 : 0.01f * v1;
            pk[r >> 1] = (unsigned)f2bf(v0) | ((unsigned)f2bf(v1) << 16);
        }
        uint4 q = make_uint4(pk[0], pk[1], pk[2], pk[3]);
        *reinterpret_cast<uint4*>(
            cbf + (size_t)n * CONV_KP + oc * 128 + owt * 16 + half * 8) = q;
    }
}

// ---------------------------------------------------------------------------
// WMMA FC kernel: D(M x NTILES*16) = act(A(M x K) @ B^T + bias)
//   Block = 8 waves = 128 rows of A x one NTILES*16-wide column group.
//   Per K-step the B tile (NTILES*16 x 32) is staged into LDS and shared by
//   all 8 waves. With USE_ASYNC_LDS the staging is GLOBAL_LOAD_ASYNC_TO_LDS
//   double-buffered (ASYNCcnt); otherwise global->reg->ds_store. A fragments
//   are register-preloaded one K-step ahead.
//   ACT: 0 = leaky_relu(0.01), 1 = tanh.  OUTBF: 1 -> bf16 out, 0 -> f32 out.
// ---------------------------------------------------------------------------
template <int NTILES, int OUTBF, int ACT>
__global__ __launch_bounds__(256)
void wmma_fc_kernel(const unsigned short* __restrict__ A, int lda,
                    const unsigned short* __restrict__ B, int ldb,
                    const float* __restrict__ bias, int nbias,
                    void* __restrict__ Cout, int ldc,
                    int K, int ngroups) {
    constexpr int ROWS   = NTILES * 16;          // B rows per K-step stage
    constexpr int CHUNKS = ROWS * 4;             // 16B chunks per stage
    constexpr int CPT    = (CHUNKS + 255) / 256; // chunks per thread (1 or 2)

    const int mb = blockIdx.x / ngroups;
    const int ng = blockIdx.x - mb * ngroups;
    const int nbase = ng * ROWS;

    const int tid  = threadIdx.x;
    const int wave = tid >> 5;
    const int lane = tid & 31;
    const int m    = lane & 15;
    const int half = lane >> 4;
    const int m_tile = mb * 8 + wave;

    v8f acc[NTILES];
#pragma unroll
    for (int t = 0; t < NTILES; ++t) acc[t] = (v8f)(0.0f);

    // A: lane holds row m of the wave's 16-row tile; k [16h..16h+15]
    const unsigned short* arow = A + (size_t)(m_tile * 16 + m) * lda + half * 16;

    const int steps = K / 32;
    union AFrag { v16bf v; v8bf h[2]; };
    AFrag a;
    a.h[0] = *reinterpret_cast<const v8bf*>(arow);
    a.h[1] = *reinterpret_cast<const v8bf*>(arow + 8);

#if USE_ASYNC_LDS
    __shared__ unsigned short Bst[2][ROWS * 32];
    __shared__ unsigned short lds_scratch[8];    // sink for inactive chunk lanes

    // chunk c -> tile row c/4, element offset (c&3)*8; invalid lanes -> scratch
    const unsigned short* bsrc[CPT];
    unsigned ldst[2][CPT];
#pragma unroll
    for (int i = 0; i < CPT; ++i) {
        int c = tid + i * 256;
        bool valid = (c < CHUNKS);
        int cc = valid ? c : 0;
        int row = cc >> 2, eo = (cc & 3) * 8;
        bsrc[i] = B + (size_t)(nbase + row) * ldb + eo;
        unsigned off = (unsigned)(row * 32 + eo);
        ldst[0][i] = valid ? lds_addr_of(&Bst[0][off]) : lds_addr_of(&lds_scratch[0]);
        ldst[1][i] = valid ? lds_addr_of(&Bst[1][off]) : lds_addr_of(&lds_scratch[0]);
    }
    // prologue: stage 0 -> buffer 0
#pragma unroll
    for (int i = 0; i < CPT; ++i) async_copy_b128(ldst[0][i], bsrc[i]);

    for (int s = 0; s < steps; ++s) {
        const int cur = s & 1;
        const int kn = (s + 1) * 32;
        if (s + 1 < steps) {
#pragma unroll
            for (int i = 0; i < CPT; ++i)
                async_copy_b128(ldst[1 - cur][i], bsrc[i] + kn);
            wait_asynccnt<CPT>();        // stage s landed; stage s+1 in flight
        } else {
            wait_asynccnt<0>();
        }
        __syncthreads();                 // all waves' stage-s copies visible

        const AFrag ac = a;
        if (s + 1 < steps) {
            a.h[0] = *reinterpret_cast<const v8bf*>(arow + kn);
            a.h[1] = *reinterpret_cast<const v8bf*>(arow + kn + 8);
        }
        if (kn + 96 < K)
            __builtin_prefetch(arow + kn + 96, 0, 3);

#pragma unroll
        for (int t = 0; t < NTILES; ++t) {
            const v16bf b = *reinterpret_cast<const v16bf*>(
                &Bst[cur][(t * 16 + m) * 32 + half * 16]);
            acc[t] = __builtin_amdgcn_wmma_f32_16x16x32_bf16(
                false, ac.v, false, b, (short)0, acc[t], false, false);
        }
        __syncthreads();                 // readers done before buf[cur] reuse
    }
#else
    __shared__ unsigned short Bst[ROWS * 32];
    const unsigned short* bsrc[CPT];
    int ldsoff[CPT];
    bool bval[CPT];
#pragma unroll
    for (int i = 0; i < CPT; ++i) {
        int c = tid + i * 256;
        bval[i] = (c < CHUNKS);
        int cc = bval[i] ? c : 0;
        int row = cc >> 2, eo = (cc & 3) * 8;
        bsrc[i]   = B + (size_t)(nbase + row) * ldb + eo;
        ldsoff[i] = row * 32 + eo;
    }
    uint4 breg[CPT];
#pragma unroll
    for (int i = 0; i < CPT; ++i)
        if (bval[i]) breg[i] = *reinterpret_cast<const uint4*>(bsrc[i]);

    for (int s = 0; s < steps; ++s) {
        __syncthreads();
#pragma unroll
        for (int i = 0; i < CPT; ++i)
            if (bval[i]) *reinterpret_cast<uint4*>(&Bst[ldsoff[i]]) = breg[i];
        const int kn = (s + 1) * 32;
        if (s + 1 < steps) {
#pragma unroll
            for (int i = 0; i < CPT; ++i)
                if (bval[i]) breg[i] = *reinterpret_cast<const uint4*>(bsrc[i] + kn);
        }
        __syncthreads();
        const AFrag ac = a;
        if (s + 1 < steps) {
            a.h[0] = *reinterpret_cast<const v8bf*>(arow + kn);
            a.h[1] = *reinterpret_cast<const v8bf*>(arow + kn + 8);
        }
#pragma unroll
        for (int t = 0; t < NTILES; ++t) {
            const v16bf b = *reinterpret_cast<const v16bf*>(
                &Bst[(t * 16 + m) * 32 + half * 16]);
            acc[t] = __builtin_amdgcn_wmma_f32_16x16x32_bf16(
                false, ac.v, false, b, (short)0, acc[t], false, false);
        }
    }
#endif

    // Epilogue: bias + activation + store. acc[t][r] -> row r+8*half, col lane&15
#pragma unroll
    for (int t = 0; t < NTILES; ++t) {
        const int col = nbase + t * 16 + m;
        const float bv = (col < nbias) ? bias[col] : 0.0f;
#pragma unroll
        for (int r = 0; r < 8; ++r) {
            const int row = m_tile * 16 + r + half * 8;
            float v = acc[t][r] + bv;
            if (ACT == 0) v = v > 0.0f ? v : 0.01f * v;
            else          v = tanhf(v);
            if (OUTBF) ((unsigned short*)Cout)[(size_t)row * ldc + col] = f2bf(v);
            else       ((float*)Cout)[(size_t)row * ldc + col] = v;
        }
    }
}

// ---------------------------------------------------------------------------
// 5) per-row: a_g = <g,hfc>, a_e = <e,hfc>; 2-way softmax; scale inputs.
//    One wave per row (wave32 butterfly reduction).
// ---------------------------------------------------------------------------
__global__ __launch_bounds__(256)
void finalize_kernel(const float* __restrict__ g, const float* __restrict__ e,
                     const float* __restrict__ hfc,
                     const float* __restrict__ ldg, const float* __restrict__ lde,
                     float* __restrict__ out_g, float* __restrict__ out_e, int n) {
    const int row  = blockIdx.x * (blockDim.x >> 5) + (threadIdx.x >> 5);
    const int lane = threadIdx.x & 31;
    if (row >= n) return;

    const float* gr = g   + (size_t)row * FC2_NP;
    const float* er = e   + (size_t)row * FC2_NP;
    const float* hr = hfc + (size_t)row * FC2_NP;
    float ag = 0.0f, ae = 0.0f;
    for (int k = lane; k < FC2_NP; k += 32) {   // pad cols are exact zeros
        const float h = hr[k];
        ag += gr[k] * h;
        ae += er[k] * h;
    }
#pragma unroll
    for (int s = 16; s > 0; s >>= 1) {
        ag += __shfl_xor(ag, s, 32);
        ae += __shfl_xor(ae, s, 32);
    }
    const float mx = fmaxf(ag, ae);
    const float w0 = __expf(ag - mx), w1 = __expf(ae - mx);
    const float inv = 1.0f / (w0 + w1);
    const float att0 = w0 * inv, att1 = w1 * inv;

    const float4* lg4 = reinterpret_cast<const float4*>(ldg + (size_t)row * 512);
    const float4* le4 = reinterpret_cast<const float4*>(lde + (size_t)row * 512);
    float4* og4 = reinterpret_cast<float4*>(out_g + (size_t)row * 512);
    float4* oe4 = reinterpret_cast<float4*>(out_e + (size_t)row * 512);
#pragma unroll
    for (int i = lane; i < 128; i += 32) {
        float4 a = lg4[i];
        float4 b = le4[i];
        og4[i] = make_float4(a.x * att0, a.y * att0, a.z * att0, a.w * att0);
        oe4[i] = make_float4(b.x * att1, b.y * att1, b.z * att1, b.w * att1);
    }
}

// ---------------------------------------------------------------------------
extern "C" void kernel_launch(void* const* d_in, const int* in_sizes, int n_in,
                              void* d_out, int out_size, void* d_ws, size_t ws_size,
                              hipStream_t stream) {
    const float* ld_gcn = (const float*)d_in[0];
    const float* ld_enc = (const float*)d_in[1];
    const int*   x      = (const int*)d_in[2];
    const int*   y      = (const int*)d_in[3];
    const float* H      = (const float*)d_in[4];
    const float* cw     = (const float*)d_in[5];
    const float* cb     = (const float*)d_in[6];
    const float* W1     = (const float*)d_in[7];
    const float* b1     = (const float*)d_in[8];
    const float* W2     = (const float*)d_in[9];
    const float* b2     = (const float*)d_in[10];
    const float* Wg     = (const float*)d_in[11];
    const float* bg     = (const float*)d_in[12];
    const float* We     = (const float*)d_in[13];
    const float* be     = (const float*)d_in[14];

    const int n = in_sizes[2];          // 8192, multiple of 128
    const int mtiles = n / 16;
    const int mblocks = mtiles / 8;     // 8 waves (m-tiles) per block

    // ---- workspace carve-up (256B aligned) ----
    char* ws = (char*)d_ws;
    size_t off = 0;
    auto carve = [&](size_t bytes) -> char* {
        char* p = ws + off;
        off = (off + bytes + 255) & ~(size_t)255;
        return p;
    };
    unsigned short* c_bf  = (unsigned short*)carve((size_t)n * CONV_KP * 2);
    unsigned short* W1p   = (unsigned short*)carve((size_t)FC1_NP * CONV_KP * 2);
    unsigned short* h1p   = (unsigned short*)carve((size_t)n * FC1_NP * 2);
    unsigned short* W2p   = (unsigned short*)carve((size_t)FC2_NP * FC1_NP * 2);
    float*          hfc   = (float*)carve((size_t)n * FC2_NP * 4);
    unsigned short* Wgp   = (unsigned short*)carve((size_t)FC2_NP * GATE_K * 2);
    unsigned short* Wep   = (unsigned short*)carve((size_t)FC2_NP * GATE_K * 2);
    float*          gbuf  = (float*)carve((size_t)n * FC2_NP * 4);
    float*          ebuf  = (float*)carve((size_t)n * FC2_NP * 4);
    unsigned short* ldgbf = (unsigned short*)carve((size_t)n * GATE_K * 2);
    unsigned short* ldebf = (unsigned short*)carve((size_t)n * GATE_K * 2);
    unsigned short* Wcp   = (unsigned short*)carve((size_t)32 * 64 * 2);

    auto blocks_for = [](int total) { return (total + 255) / 256; };

    // ---- 0) converts / padding ----
    pad_w1_kernel<<<blocks_for(FC1_NP * CONV_KP), 256, 0, stream>>>(W1p, W1);
    pad_bf16_kernel<<<blocks_for(FC2_NP * FC1_NP), 256, 0, stream>>>(W2p, W2, FC2_NP, FC1_NP, 100, 1000);
    pad_bf16_kernel<<<blocks_for(FC2_NP * GATE_K), 256, 0, stream>>>(Wgp, Wg, FC2_NP, GATE_K, 100, GATE_K);
    pad_bf16_kernel<<<blocks_for(FC2_NP * GATE_K), 256, 0, stream>>>(Wep, We, FC2_NP, GATE_K, 100, GATE_K);
    pad_bf16_kernel<<<blocks_for(n * GATE_K), 256, 0, stream>>>(ldgbf, ld_gcn, n, GATE_K, n, GATE_K);
    pad_bf16_kernel<<<blocks_for(n * GATE_K), 256, 0, stream>>>(ldebf, ld_enc, n, GATE_K, n, GATE_K);
    // conv_w (32,1,2,25) is 32 rows of 50 contiguous floats -> pad to 32x64
    pad_bf16_kernel<<<blocks_for(32 * 64), 256, 0, stream>>>(Wcp, cw, 32, 64, 32, 50);

    // ---- 1) embedding + im2col + conv (WMMA) + lrelu ----
    conv_embed_wmma<<<n, 256, 0, stream>>>(x, y, H, Wcp, cb, c_bf);

    // ---- 2) FC1: block = 128 rows x 64 cols, B via async LDS; 16 N-groups ----
    wmma_fc_kernel<4, 1, 0><<<mblocks * (FC1_NP / 64), 256, 0, stream>>>(
        c_bf, CONV_KP, W1p, CONV_KP, b1, 1000, h1p, FC1_NP, CONV_KP, FC1_NP / 64);

    // ---- 3) FC2: block = 128 rows x 112 cols ----
    wmma_fc_kernel<7, 0, 0><<<mblocks, 256, 0, stream>>>(
        h1p, FC1_NP, W2p, FC1_NP, b2, 100, hfc, FC2_NP, FC1_NP, 1);

    // ---- 4) gating GEMMs with tanh ----
    wmma_fc_kernel<7, 0, 1><<<mblocks, 256, 0, stream>>>(
        ldgbf, GATE_K, Wgp, GATE_K, bg, 100, gbuf, FC2_NP, GATE_K, 1);
    wmma_fc_kernel<7, 0, 1><<<mblocks, 256, 0, stream>>>(
        ldebf, GATE_K, Wep, GATE_K, be, 100, ebuf, FC2_NP, GATE_K, 1);

    // ---- 5) dots + softmax + scale ----
    float* out_g = (float*)d_out;
    float* out_e = out_g + (size_t)n * 512;
    finalize_kernel<<<(n + 7) / 8, 256, 0, stream>>>(gbuf, ebuf, hfc, ld_gcn, ld_enc,
                                                     out_g, out_e, n);
}
